// PointNet2SemSegSSG_66795331388117
// MI455X (gfx1250) — compile-verified, matrix operand order checked by reading, not run on hardware
//
#include <hip/hip_runtime.h>

typedef _Float16 f16;
typedef __attribute__((ext_vector_type(16))) _Float16 v16h;
typedef __attribute__((ext_vector_type(8)))  float    v8f;
typedef unsigned int u32;
typedef __attribute__((ext_vector_type(4))) u32 v4u;
typedef __attribute__((ext_vector_type(8))) int v8i;
typedef __attribute__((ext_vector_type(4))) int v4i;

#define BATCH 4
#define NPTS  16384
#define MAX_KP 768   // largest GEMM K dimension (FP4 concat: 256+512)

static inline int ceil32(int k) { return (k + 31) & ~31; }

// ---------------------------------------------------------------- kernels ---

// points (B,3,N) -> xyz (B,N,3)
__global__ void k_transpose(const float* __restrict__ pts, float* __restrict__ xyz, int total) {
  int i = blockIdx.x * blockDim.x + threadIdx.x;
  if (i >= total) return;
  int b = i / NPTS, n = i - b * NPTS;
  const float* src = pts + (size_t)b * 3 * NPTS + n;
  float* dst = xyz + (size_t)i * 3;
  dst[0] = src[0];
  dst[1] = src[NPTS];
  dst[2] = src[2 * NPTS];
}

// farthest point sampling, one block per batch (sequential argmax chain)
__global__ void k_fps(const float* __restrict__ xyz, int n, int npoint,
                      int* __restrict__ idx, float* __restrict__ dmin_ws) {
  int b = blockIdx.x, tid = threadIdx.x;
  const float* x = xyz + (size_t)b * n * 3;
  float* dmin = dmin_ws + (size_t)b * n;
  __shared__ float smax[256];
  __shared__ int   simax[256];
  __shared__ int   sfar;
  for (int i = tid; i < n; i += 256) dmin[i] = 1e10f;
  if (tid == 0) sfar = 0;
  __syncthreads();
  for (int j = 0; j < npoint; ++j) {
    int far = sfar;
    if (tid == 0) idx[b * npoint + j] = far;
    float cx = x[far * 3 + 0], cy = x[far * 3 + 1], cz = x[far * 3 + 2];
    float bv = -1.0f; int bi = 0;
    for (int i = tid; i < n; i += 256) {
      float dx = x[i * 3 + 0] - cx, dy = x[i * 3 + 1] - cy, dz = x[i * 3 + 2] - cz;
      float d = dx * dx + dy * dy + dz * dz;
      float dm = fminf(dmin[i], d);
      dmin[i] = dm;
      if (dm > bv) { bv = dm; bi = i; }
    }
    smax[tid] = bv; simax[tid] = bi;
    __syncthreads();
    for (int s = 128; s > 0; s >>= 1) {
      if (tid < s) {
        float ov = smax[tid + s]; int oi = simax[tid + s];
        if (ov > smax[tid] || (ov == smax[tid] && oi < simax[tid])) {
          smax[tid] = ov; simax[tid] = oi;
        }
      }
      __syncthreads();
    }
    if (tid == 0) sfar = simax[0];
    __syncthreads();
  }
}

__global__ void k_gather_xyz(const float* __restrict__ xyz, const int* __restrict__ idx,
                             float* __restrict__ out, int n, int total) {
  int i = blockIdx.x * blockDim.x + threadIdx.x;   // over B*S
  if (i >= total) return;
  int S = total / BATCH;
  int b = i / S;
  int gi = idx[i];
  const float* p = xyz + (size_t)(b * n + gi) * 3;
  out[(size_t)i * 3 + 0] = p[0];
  out[(size_t)i * 3 + 1] = p[1];
  out[(size_t)i * 3 + 2] = p[2];
}

// first-K-in-index-order ball query (matches top_k over masked index trick)
__global__ void k_ballquery(const float* __restrict__ xyz, const float* __restrict__ nxyz,
                            int* __restrict__ gidx, int n, int S, float r2, int K, int total) {
  int row = blockIdx.x * blockDim.x + threadIdx.x;
  if (row >= total) return;
  int b = row / S;
  const float* c = nxyz + (size_t)row * 3;
  float cx = c[0], cy = c[1], cz = c[2];
  const float* x = xyz + (size_t)b * n * 3;
  int* g = gidx + (size_t)row * K;
  int cnt = 0;
  for (int i = 0; i < n && cnt < K; ++i) {
    float dx = x[i * 3] - cx, dy = x[i * 3 + 1] - cy, dz = x[i * 3 + 2] - cz;
    if (dx * dx + dy * dy + dz * dz <= r2) g[cnt++] = i;
  }
  int first = (cnt > 0) ? g[0] : 0;
  for (int k = cnt; k < K; ++k) g[k] = first;
}

// build grouped [rel_xyz(3) | feats(Cf)] rows, padded to Kp, as f16 GEMM operand
__global__ void k_group(const float* __restrict__ xyz, const float* __restrict__ nxyz,
                        const float* __restrict__ feats, const int* __restrict__ gidx,
                        f16* __restrict__ A, int n, int S, int K, int Cf, int Kp, int total) {
  int row = blockIdx.x * blockDim.x + threadIdx.x;  // (b*S+s)*K + k
  if (row >= total) return;
  int bs = row / K;
  int b = bs / S;
  int gi = gidx[row];
  const float* p = xyz + (size_t)(b * n + gi) * 3;
  const float* c = nxyz + (size_t)bs * 3;
  f16* out = A + (size_t)row * Kp;
  out[0] = (f16)(p[0] - c[0]);
  out[1] = (f16)(p[1] - c[1]);
  out[2] = (f16)(p[2] - c[2]);
  const float* f = feats + (size_t)(b * n + gi) * Cf;
  for (int j = 0; j < Cf; ++j) out[3 + j] = (f16)f[j];
  for (int j = 3 + Cf; j < Kp; ++j) out[j] = (f16)0.f;
}

// weights (N,K) f32 -> (N,Kp) f16, zero padded
__global__ void k_cvtw(const float* __restrict__ W, f16* __restrict__ Wh,
                       int K, int Kp, int total) {
  int i = blockIdx.x * blockDim.x + threadIdx.x;
  if (i >= total) return;
  int n = i / Kp, k = i - n * Kp;
  Wh[i] = (k < K) ? (f16)W[(size_t)n * K + k] : (f16)0.f;
}

// Y[M,N] = A[M,Kp] * W[N,Kp]^T
// grid = (N/16, M/64), block = (32,4): one wave32 per 16x16 C tile.
// All 4 waves share the same weight tile (same blockIdx.x), so the 16xKp
// B tile is staged once per block into LDS by the Tensor Data Mover and
// read back via ds loads; A streams from global into WMMA fragments.
__global__ void k_gemm(const f16* __restrict__ A, const f16* __restrict__ Bw,
                       float* __restrict__ Y, int N, int Kp) {
  __shared__ __align__(16) f16 Bs[16 * MAX_KP];   // 24 KB
  int lane = threadIdx.x;
  int wy = threadIdx.y;
  int l15 = lane & 15, hi = lane >> 4;
  int tn = blockIdx.x;
  int tm = blockIdx.y * blockDim.y + wy;

  // ---- stage the 16 x Kp weight tile into LDS (wave 0 only) ----
  if (wy == 0) {
#if __has_builtin(__builtin_amdgcn_tensor_load_to_lds) && __has_builtin(__builtin_amdgcn_s_wait_tensorcnt)
    // Tensor DMA descriptor (CDNA5 ISA): 2D tile, 2-byte elems,
    // tile_dim = (Kp, 16), tensor_dim = (Kp, 16), dim0_stride = Kp.
    unsigned long long gaddr = (unsigned long long)(const void*)(Bw + (size_t)tn * 16 * Kp);
    u32 ldsoff = (u32)(unsigned long long)(const void*)&Bs[0]; // low 32b of flat = LDS offset
    v4u g0;
    g0[0] = 1u;                                          // count=1 (valid), user mode
    g0[1] = ldsoff;                                      // lds_addr
    g0[2] = (u32)gaddr;                                  // global_addr[31:0]
    g0[3] = (u32)((gaddr >> 32) & 0x1FFFFFFu) | (2u << 30); // addr[56:32] | type=2
    v8i g1;
    g1[0] = (int)(1u << 16);                             // data_size=1 (2 bytes)
    g1[1] = (int)(((u32)Kp & 0xFFFFu) << 16);            // tensor_dim0[15:0] @ bits 63:48
    g1[2] = (int)((((u32)Kp >> 16) & 0xFFFFu) | (16u << 16)); // dim0[31:16] | tensor_dim1=16
    g1[3] = (int)(((u32)Kp & 0xFFFFu) << 16);            // tile_dim0=Kp @ bits 127:112
    g1[4] = 16;                                          // tile_dim1=16, tile_dim2=0
    g1[5] = Kp;                                          // tensor_dim0_stride[31:0]
    g1[6] = 0;                                           // stride0[47:32]=0, stride1 lo=0
    g1[7] = 0;
    v4i gz4 = {0, 0, 0, 0};
    v8i gz8 = {0, 0, 0, 0, 0, 0, 0, 0};
    // 6-arg form (amdgpu-toolchain clang-23): (v4u, v8i, v4i, v4i, v8i, cpol)
    __builtin_amdgcn_tensor_load_to_lds(g0, g1, gz4, gz4, gz8, 0);
    __builtin_amdgcn_s_wait_tensorcnt(0);
#else
    // fallback: cooperative staged copy (64-bit chunks, 16*Kp halfs)
    const unsigned long long* src =
        (const unsigned long long*)(Bw + (size_t)tn * 16 * Kp);
    unsigned long long* dst = (unsigned long long*)&Bs[0];
    int nq = 16 * Kp / 4;
    for (int i = lane; i < nq; i += 32) dst[i] = src[i];
#endif
  }
  __syncthreads();

  const f16* Ar = A + (size_t)(tm * 16 + l15) * Kp;
  const f16* Br = Bs + (size_t)l15 * Kp;   // LDS
  v8f c = {};
  for (int k0 = 0; k0 < Kp; k0 += 32) {
    v16h a, bb;
#pragma unroll
    for (int v = 0; v < 8; ++v) {
      // ISA 7.12.2 16-bit A/B layout: lanes 0-15 hold K 0-7 (v0-3) & 16-23 (v4-7),
      // lanes 16-31 hold K 8-15 & 24-31; two f16 per VGPR.
      int k = k0 + ((v < 4) ? 2 * v : 16 + 2 * (v - 4)) + hi * 8;
      a[2 * v]      = Ar[k];
      a[2 * v + 1]  = Ar[k + 1];
      bb[2 * v]     = Br[k];
      bb[2 * v + 1] = Br[k + 1];
    }
    c = __builtin_amdgcn_wmma_f32_16x16x32_f16(false, a, false, bb,
                                               (short)0, c, false, false);
  }
  // C/D layout: VGPR v -> row v (lanes 0-15) or row v+8 (lanes 16-31), col = lane&15
  float* Yr = Y + (size_t)(tm * 16 + hi * 8) * N + tn * 16 + l15;
#pragma unroll
  for (int v = 0; v < 8; ++v) Yr[(size_t)v * N] = c[v];
}

// per-channel batch stats over all M rows -> scale/shift for fused BN
__global__ void k_bnstats(const float* __restrict__ Y, const float* __restrict__ g,
                          const float* __restrict__ b, float* __restrict__ scale,
                          float* __restrict__ shift, int M, int N) {
  int c = blockIdx.x, tid = threadIdx.x;
  __shared__ float ssum[256], ssq[256];
  float s = 0.f, q = 0.f;
  for (int i = tid; i < M; i += 256) {
    float v = Y[(size_t)i * N + c];
    s += v; q += v * v;
  }
  ssum[tid] = s; ssq[tid] = q;
  __syncthreads();
  for (int st = 128; st > 0; st >>= 1) {
    if (tid < st) { ssum[tid] += ssum[tid + st]; ssq[tid] += ssq[tid + st]; }
    __syncthreads();
  }
  if (tid == 0) {
    float mu = ssum[0] / (float)M;
    float var = ssq[0] / (float)M - mu * mu;
    float sc = g[c] * rsqrtf(var + 1e-5f);
    scale[c] = sc;
    shift[c] = b[c] - mu * sc;
  }
}

__global__ void k_bnrelu_f16(const float* __restrict__ Y, const float* __restrict__ scale,
                             const float* __restrict__ shift, f16* __restrict__ X,
                             int N, int total) {
  int i = blockIdx.x * blockDim.x + threadIdx.x;
  if (i >= total) return;
  int c = i % N;
  X[i] = (f16)fmaxf(0.f, Y[i] * scale[c] + shift[c]);
}

__global__ void k_bnrelu_f32(const float* __restrict__ Y, const float* __restrict__ scale,
                             const float* __restrict__ shift, float* __restrict__ X,
                             int N, int total) {
  int i = blockIdx.x * blockDim.x + threadIdx.x;
  if (i >= total) return;
  int c = i % N;
  X[i] = fmaxf(0.f, Y[i] * scale[c] + shift[c]);
}

// max over nsample (rows are (bs*K+k)), f16 activations -> f32 level features
__global__ void k_maxpool(const f16* __restrict__ X, float* __restrict__ F,
                          int K, int C, int total) {
  int i = blockIdx.x * blockDim.x + threadIdx.x;   // over (B*S)*C
  if (i >= total) return;
  int bs = i / C, c = i - bs * C;
  const f16* src = X + (size_t)bs * K * C + c;
  float m = -1e30f;
  for (int k = 0; k < K; ++k) m = fmaxf(m, (float)src[(size_t)k * C]);
  F[i] = m;
}

// 3-NN inverse-distance interpolation + concat [p1 | interp] padded to Kp (f16)
__global__ void k_fpinterp(const float* __restrict__ xyz1, const float* __restrict__ xyz2,
                           const float* __restrict__ p1f, const float* __restrict__ f2,
                           f16* __restrict__ A, int n1, int n2, int C1, int C2,
                           int Kp, int total) {
  int r = blockIdx.x * blockDim.x + threadIdx.x;   // b*n1 + i
  if (r >= total) return;
  int b = r / n1;
  const float* p = xyz1 + (size_t)r * 3;
  float px = p[0], py = p[1], pz = p[2];
  const float* x2 = xyz2 + (size_t)b * n2 * 3;
  float d0 = 1e30f, d1 = 1e30f, d2 = 1e30f;
  int i0 = 0, i1 = 0, i2 = 0;
  for (int j = 0; j < n2; ++j) {
    float dx = x2[j * 3] - px, dy = x2[j * 3 + 1] - py, dz = x2[j * 3 + 2] - pz;
    float d = dx * dx + dy * dy + dz * dz;
    if (d < d0)      { d2 = d1; i2 = i1; d1 = d0; i1 = i0; d0 = d; i0 = j; }
    else if (d < d1) { d2 = d1; i2 = i1; d1 = d;  i1 = j; }
    else if (d < d2) { d2 = d;  i2 = j; }
  }
  float w0 = 1.f / (d0 + 1e-8f), w1 = 1.f / (d1 + 1e-8f), w2 = 1.f / (d2 + 1e-8f);
  float ws = w0 + w1 + w2;
  w0 /= ws; w1 /= ws; w2 /= ws;
  f16* out = A + (size_t)r * Kp;
  const float* pf = p1f + (size_t)r * C1;
  for (int c = 0; c < C1; ++c) out[c] = (f16)pf[c];
  const float* base = f2 + (size_t)b * n2 * C2;
  const float* fA = base + (size_t)i0 * C2;
  const float* fB = base + (size_t)i1 * C2;
  const float* fC = base + (size_t)i2 * C2;
  for (int c = 0; c < C2; ++c)
    out[C1 + c] = (f16)(w0 * fA[c] + w1 * fB[c] + w2 * fC[c]);
  for (int c = C1 + C2; c < Kp; ++c) out[c] = (f16)0.f;
}

// final BN+ReLU and transpose (B,N,C) -> (B,C,N) into d_out
__global__ void k_out(const float* __restrict__ Y, const float* __restrict__ scale,
                      const float* __restrict__ shift, float* __restrict__ out,
                      int C, int total) {
  int idx = blockIdx.x * blockDim.x + threadIdx.x;
  if (idx >= total) return;
  int row = idx / C, c = idx - row * C;
  int b = row / NPTS, i = row - b * NPTS;
  out[((size_t)b * C + c) * NPTS + i] = fmaxf(0.f, Y[idx] * scale[c] + shift[c]);
}

// ---------------------------------------------------------------- host ------

struct LayerP { const float* W; const float* g; const float* b; int K; int N; };

static void run_mlp(const LayerP* L, int nL, int M, int Kp0,
                    f16* Abuf, float* Ybuf, f16* Wh, float* scale, float* shift,
                    float* f32out, hipStream_t s) {
  int Kp = Kp0;
  for (int l = 0; l < nL; ++l) {
    int K = L[l].K, N = L[l].N;
    int nW = N * Kp;
    k_cvtw<<<(nW + 255) / 256, 256, 0, s>>>(L[l].W, Wh, K, Kp, nW);
    dim3 g(N / 16, M / 64), t(32, 4);
    k_gemm<<<g, t, 0, s>>>(Abuf, Wh, Ybuf, N, Kp);
    k_bnstats<<<N, 256, 0, s>>>(Ybuf, L[l].g, L[l].b, scale, shift, M, N);
    int tot = M * N;
    if (l == nL - 1 && f32out)
      k_bnrelu_f32<<<(tot + 255) / 256, 256, 0, s>>>(Ybuf, scale, shift, f32out, N, tot);
    else
      k_bnrelu_f16<<<(tot + 255) / 256, 256, 0, s>>>(Ybuf, scale, shift, Abuf, N, tot);
    Kp = N;   // all Cout are multiples of 32
  }
}

static void run_sa(const float* xyz_in, const float* feat_in, int n, int Cf,
                   int S, float radius, int K, const LayerP* L, int nL,
                   float* xyz_out, float* feat_out,
                   int* fpsidx, int* gidx, float* dminws,
                   f16* Abuf, float* Ybuf, f16* Wh, float* scale, float* shift,
                   hipStream_t s) {
  k_fps<<<BATCH, 256, 0, s>>>(xyz_in, n, S, fpsidx, dminws);
  int bs = BATCH * S;
  k_gather_xyz<<<(bs + 255) / 256, 256, 0, s>>>(xyz_in, fpsidx, xyz_out, n, bs);
  k_ballquery<<<(bs + 255) / 256, 256, 0, s>>>(xyz_in, xyz_out, gidx, n, S,
                                               radius * radius, K, bs);
  int M = bs * K;
  int Kp0 = ceil32(3 + Cf);
  k_group<<<(M + 255) / 256, 256, 0, s>>>(xyz_in, xyz_out, feat_in, gidx, Abuf,
                                          n, S, K, Cf, Kp0, M);
  run_mlp(L, nL, M, Kp0, Abuf, Ybuf, Wh, scale, shift, nullptr, s);
  int Cout = L[nL - 1].N;
  int tot = bs * Cout;
  k_maxpool<<<(tot + 255) / 256, 256, 0, s>>>(Abuf, feat_out, K, Cout, tot);
}

static void run_fp(const float* xyz1, const float* xyz2, const float* p1f, int C1,
                   const float* f2, int C2, int n1, int n2,
                   const LayerP* L, int nL, float* fout,
                   f16* Abuf, float* Ybuf, f16* Wh, float* scale, float* shift,
                   hipStream_t s) {
  int M = BATCH * n1;
  int Kp0 = ceil32(C1 + C2);
  k_fpinterp<<<(M + 255) / 256, 256, 0, s>>>(xyz1, xyz2, p1f, f2, Abuf,
                                             n1, n2, C1, C2, Kp0, M);
  run_mlp(L, nL, M, Kp0, Abuf, Ybuf, Wh, scale, shift, fout, s);
}

extern "C" void kernel_launch(void* const* d_in, const int* in_sizes, int n_in,
                              void* d_out, int out_size, void* d_ws, size_t ws_size,
                              hipStream_t stream) {
  (void)in_sizes; (void)out_size; (void)ws_size;
  if (n_in < 67) return;

  // --- input map: [0]=points; params pytree (sorted keys): fc(3), fp(27), sa(36)
  auto P = [&](int i) { return (const float*)d_in[i]; };
  const float* points = P(0);
  const float* fcW = P(1); const float* fcg = P(2); const float* fcb = P(3);
  LayerP fp0[3] = { {P(4),P(5),P(6),131,128}, {P(7),P(8),P(9),128,128}, {P(10),P(11),P(12),128,128} };
  LayerP fp1[2] = { {P(13),P(14),P(15),320,256}, {P(16),P(17),P(18),256,128} };
  LayerP fp2[2] = { {P(19),P(20),P(21),384,256}, {P(22),P(23),P(24),256,256} };
  LayerP fp3[2] = { {P(25),P(26),P(27),768,256}, {P(28),P(29),P(30),256,256} };
  LayerP sa1[3] = { {P(31),P(32),P(33),6,32},   {P(34),P(35),P(36),32,32},   {P(37),P(38),P(39),32,64} };
  LayerP sa2[3] = { {P(40),P(41),P(42),67,64},  {P(43),P(44),P(45),64,64},   {P(46),P(47),P(48),64,128} };
  LayerP sa3[3] = { {P(49),P(50),P(51),131,128},{P(52),P(53),P(54),128,128}, {P(55),P(56),P(57),128,256} };
  LayerP sa4[3] = { {P(58),P(59),P(60),259,256},{P(61),P(62),P(63),256,256}, {P(64),P(65),P(66),256,512} };

  // --- workspace layout (bump allocator, 256B aligned) ---
  char* base = (char*)d_ws;
  size_t off = 0;
  auto alloc = [&](size_t bytes) -> void* {
    void* p = base + off;
    off = (off + bytes + 255) & ~(size_t)255;
    return p;
  };
  float* xyz0   = (float*)alloc((size_t)BATCH * NPTS * 3 * 4);
  float* dminws = (float*)alloc((size_t)BATCH * NPTS * 4);
  int*   fpsidx = (int*)  alloc((size_t)BATCH * 1024 * 4);
  int*   gidx   = (int*)  alloc((size_t)BATCH * 1024 * 32 * 4);
  float* xyz1   = (float*)alloc((size_t)BATCH * 1024 * 3 * 4);
  float* xyz2   = (float*)alloc((size_t)BATCH * 256 * 3 * 4);
  float* xyz3   = (float*)alloc((size_t)BATCH * 64 * 3 * 4);
  float* xyz4   = (float*)alloc((size_t)BATCH * 16 * 3 * 4);
  float* f1     = (float*)alloc((size_t)BATCH * 1024 * 64 * 4);
  float* f2     = (float*)alloc((size_t)BATCH * 256 * 128 * 4);
  float* f3     = (float*)alloc((size_t)BATCH * 64 * 256 * 4);
  float* f4     = (float*)alloc((size_t)BATCH * 16 * 512 * 4);
  float* fpo3   = (float*)alloc((size_t)BATCH * 64 * 256 * 4);
  float* fpo2   = (float*)alloc((size_t)BATCH * 256 * 256 * 4);
  float* fpo1   = (float*)alloc((size_t)BATCH * 1024 * 128 * 4);
  f16*   Wh     = (f16*)  alloc((size_t)256 * 768 * 2);
  float* scale  = (float*)alloc(512 * 4);
  float* shift  = (float*)alloc(512 * 4);
  f16*   Abuf   = (f16*)  alloc((size_t)BATCH * NPTS * 160 * 2);   // max A (f16)
  float* Ybuf   = (float*)alloc((size_t)BATCH * NPTS * 128 * 4);   // max Y (f32)

  // --- pipeline ---
  int totp = BATCH * NPTS;
  k_transpose<<<(totp + 255) / 256, 256, 0, stream>>>(points, xyz0, totp);

  run_sa(xyz0, xyz0, NPTS, 3,   1024, 0.1f, 32, sa1, 3, xyz1, f1,
         fpsidx, gidx, dminws, Abuf, Ybuf, Wh, scale, shift, stream);
  run_sa(xyz1, f1,   1024, 64,  256,  0.2f, 32, sa2, 3, xyz2, f2,
         fpsidx, gidx, dminws, Abuf, Ybuf, Wh, scale, shift, stream);
  run_sa(xyz2, f2,   256,  128, 64,   0.4f, 32, sa3, 3, xyz3, f3,
         fpsidx, gidx, dminws, Abuf, Ybuf, Wh, scale, shift, stream);
  run_sa(xyz3, f3,   64,   256, 16,   0.8f, 32, sa4, 3, xyz4, f4,
         fpsidx, gidx, dminws, Abuf, Ybuf, Wh, scale, shift, stream);

  run_fp(xyz3, xyz4, f3,   256, f4,   512, 64,   16,   fp3, 2, fpo3,
         Abuf, Ybuf, Wh, scale, shift, stream);
  run_fp(xyz2, xyz3, f2,   128, fpo3, 256, 256,  64,   fp2, 2, fpo2,
         Abuf, Ybuf, Wh, scale, shift, stream);
  run_fp(xyz1, xyz2, f1,   64,  fpo2, 256, 1024, 256,  fp1, 2, fpo1,
         Abuf, Ybuf, Wh, scale, shift, stream);
  run_fp(xyz0, xyz1, xyz0, 3,   fpo1, 128, NPTS, 1024, fp0, 3, nullptr,
         Abuf, Ybuf, Wh, scale, shift, stream);   // final f16 acts stay in Abuf

  // --- final FC (128x128) + BN + ReLU + transpose to (B,128,N) ---
  {
    int M = BATCH * NPTS, N = 128, Kp = 128;
    k_cvtw<<<(N * Kp + 255) / 256, 256, 0, stream>>>(fcW, Wh, 128, Kp, N * Kp);
    dim3 g(N / 16, M / 64), t(32, 4);
    k_gemm<<<g, t, 0, stream>>>(Abuf, Wh, Ybuf, N, Kp);
    k_bnstats<<<N, 256, 0, stream>>>(Ybuf, fcg, fcb, scale, shift, M, N);
    int tot = M * N;
    k_out<<<(tot + 255) / 256, 256, 0, stream>>>(Ybuf, scale, shift,
                                                 (float*)d_out, N, tot);
  }
}
